// GAT_43508018709271
// MI455X (gfx1250) — matrix operand order repaired and checked by reading, not compiled, not run on hardware
//
#include <hip/hip_runtime.h>
#include <math.h>

typedef __attribute__((ext_vector_type(16))) _Float16 v16h;
typedef __attribute__((ext_vector_type(8)))  float    v8f;

// ---------------- helpers ----------------

__device__ __forceinline__ float lrelu(float v, float s) { return v > 0.f ? v : s * v; }

// order-preserving float -> uint key for atomicMax-based segment max
__device__ __forceinline__ unsigned fkey(float x) {
    unsigned b = __float_as_uint(x);
    return (b & 0x80000000u) ? ~b : (b | 0x80000000u);
}
__device__ __forceinline__ float fdec(unsigned k) {
    unsigned b = (k & 0x80000000u) ? (k & 0x7FFFFFFFu) : ~k;
    return __uint_as_float(b);
}

// edge e in [0, Etot): first E edges from edge_index (row-major [2,E]), rest self-loops
__device__ __forceinline__ void edge_nodes(const int* __restrict__ ei, int E, int e,
                                           int& s, int& d) {
    if (e < E) { s = ei[e]; d = ei[E + e]; }
    else       { s = e - E; d = e - E; }
}

// ---------------- WMMA GEMM: Out[M,N] = X[M,K] @ W[K,N], f32 io, f16 WMMA ----------------
// Block: 256 threads = 8 waves as 2(M) x 4(N); each wave computes 32x32 -> 4 WMMA/K-step.
// Tile: BM=64, BN=128, BK=32. N (=Fout) must be a multiple of 128 (true: 1024/512/128).
// LDS tiles are stored PRE-SWIZZLED into WMMA fragment order (each lane's v16h contiguous,
// read as 2x ds_load_b128), and DOUBLE-BUFFERED so next-tile staging overlaps the WMMAs.

#define BM 64
#define BN 128
#define BK 32
#define FPAD 24   // halves per (tile,lane) chunk: 16 data + 8 pad (48B lane stride, 16B aligned)
#define SZA ((BM / 16) * 32 * FPAD)
#define SZB ((BN / 16) * 32 * FPAD)

__global__ __launch_bounds__(256)
void gemm_wmma_kernel(const float* __restrict__ X, const float* __restrict__ W,
                      float* __restrict__ Out, int M, int K, int N) {
    __shared__ _Float16 sA[2 * SZA];
    __shared__ _Float16 sB[2 * SZB];

    const int tid    = threadIdx.x;
    const int lane   = tid & 31;
    const int wave   = tid >> 5;
    const int wm     = wave & 1;   // 2 M-supertiles of 32 rows
    const int wn     = wave >> 1;  // 4 N-supertiles of 32 cols
    const int blockM = blockIdx.y * BM;
    const int blockN = blockIdx.x * BN;

    const int half16 = lane >> 4;
    const int mloc   = lane & 15;

    v8f acc[2][2];
    #pragma unroll
    for (int p = 0; p < 2; ++p)
        #pragma unroll
        for (int q = 0; q < 2; ++q)
            acc[p][q] = (v8f){0.f, 0.f, 0.f, 0.f, 0.f, 0.f, 0.f, 0.f};

    // staging coordinates (fixed per thread)
    const int ar  = tid >> 2;          // 0..63   A row
    const int ac0 = (tid & 3) * 8;     // A col chunk
    const int br  = tid >> 3;          // 0..31   B row (K)
    const int bc0 = (tid & 7) * 16;    // B col chunk

    int grA = blockM + ar;
    if (grA > M - 1) grA = M - 1;
    const float* rowA = X + (size_t)grA * K;
    const int tA = ar >> 4, mA = ar & 15;

    // stage tiles for K-chunk kk into buffer bo (0/1); prefetch 2 chunks ahead.
    auto stage = [&](int kk, int bo) {
        const int aoff = bo * SZA, boff = bo * SZB;
        // ---- A (BM x BK): fast vector path when the whole chunk is K-interior ----
        float av[8];
        if (kk + BK <= K) {            // uniform branch (K multiple of BK: layers 2,3)
            float4 f0 = *(const float4*)(rowA + kk + ac0);
            float4 f1 = *(const float4*)(rowA + kk + ac0 + 4);
            av[0] = f0.x; av[1] = f0.y; av[2] = f0.z; av[3] = f0.w;
            av[4] = f1.x; av[5] = f1.y; av[6] = f1.z; av[7] = f1.w;
        } else {                       // clamped scalar path (layer-1 K=11 only)
            #pragma unroll
            for (int u = 0; u < 8; ++u) {
                int gc = kk + ac0 + u;
                int ci = gc < K ? gc : 0;
                float v = rowA[ci];
                av[u] = gc < K ? v : 0.f;
            }
        }
        #pragma unroll
        for (int u = 0; u < 8; ++u) {
            int c   = ac0 + u;
            int ln  = mA + (((c >> 3) & 1) << 4);
            int pos = (c & 7) + (((c >> 4) & 1) << 3);
            sA[aoff + (tA * 32 + ln) * FPAD + pos] = (_Float16)av[u];
        }
        {
            int pk = kk + BK + ac0;
            if (pk > K - 1) pk = K - 1;
            __builtin_prefetch(rowA + pk, 0, 0);
        }
        // ---- B (BK x BN) via aligned float4 ----
        {
            int grow = kk + br;
            int rc   = grow < K ? grow : K - 1;
            bool ok  = grow < K;
            const float* row = W + (size_t)rc * N + blockN;
            #pragma unroll
            for (int v4 = 0; v4 < 4; ++v4) {
                float4 f = *(const float4*)(row + bc0 + 4 * v4);
                float vv[4] = {f.x, f.y, f.z, f.w};
                #pragma unroll
                for (int u = 0; u < 4; ++u) {
                    int c = bc0 + 4 * v4 + u;
                    float val = ok ? vv[u] : 0.f;
                    int ln  = (c & 15) + (((br >> 3) & 1) << 4);
                    int pos = (br & 7) + (((br >> 4) & 1) << 3);
                    sB[boff + ((c >> 4) * 32 + ln) * FPAD + pos] = (_Float16)val;
                }
            }
            int pr = kk + BK + br;
            if (pr > K - 1) pr = K - 1;
            __builtin_prefetch(W + (size_t)pr * N + blockN + bc0, 0, 0);
        }
    };

    stage(0, 0);
    int cur = 0;
    for (int k0 = 0; k0 < K; k0 += BK) {
        __syncthreads();
        if (k0 + BK < K) stage(k0 + BK, cur ^ 1);

        const int ao = cur * SZA, bo = cur * SZB;
        v16h a0 = *(const v16h*)&sA[ao + ((2 * wm + 0) * 32 + lane) * FPAD];
        v16h a1 = *(const v16h*)&sA[ao + ((2 * wm + 1) * 32 + lane) * FPAD];
        v16h b0 = *(const v16h*)&sB[bo + ((2 * wn + 0) * 32 + lane) * FPAD];
        v16h b1 = *(const v16h*)&sB[bo + ((2 * wn + 1) * 32 + lane) * FPAD];

        acc[0][0] = __builtin_amdgcn_wmma_f32_16x16x32_f16(false, a0, false, b0, (short)0, acc[0][0], false, false);
        acc[0][1] = __builtin_amdgcn_wmma_f32_16x16x32_f16(false, a0, false, b1, (short)0, acc[0][1], false, false);
        acc[1][0] = __builtin_amdgcn_wmma_f32_16x16x32_f16(false, a1, false, b0, (short)0, acc[1][0], false, false);
        acc[1][1] = __builtin_amdgcn_wmma_f32_16x16x32_f16(false, a1, false, b1, (short)0, acc[1][1], false, false);
        cur ^= 1;
    }

    // ---- store: C/D layout VGPR j -> row = tile*16 + half16*8 + j, col = tile*16 + mloc ----
    const bool full = (blockM + BM) <= M;   // uniform per block: fast path for interior tiles
    #pragma unroll
    for (int p = 0; p < 2; ++p) {
        #pragma unroll
        for (int q = 0; q < 2; ++q) {
            int col  = blockN + (2 * wn + q) * 16 + mloc;
            int row0 = blockM + (2 * wm + p) * 16 + half16 * 8;
            float* op = Out + (size_t)row0 * N + col;
            if (full) {
                #pragma unroll
                for (int j = 0; j < 8; ++j)
                    op[(size_t)j * N] = acc[p][q][j];
            } else {
                #pragma unroll
                for (int j = 0; j < 8; ++j)
                    if (row0 + j < M) op[(size_t)j * N] = acc[p][q][j];
            }
        }
    }
}

// ---------------- attention logits: one wave per (n,h), coalesced + shfl reduce ----------------

__global__ __launch_bounds__(256)
void attn_logits_kernel(const float* __restrict__ H, const float* __restrict__ a_s,
                        const float* __restrict__ a_d, float* __restrict__ als,
                        float* __restrict__ ald, int Nn, int Hh, int C) {
    long wid = ((long)blockIdx.x * blockDim.x + threadIdx.x) >> 5;
    int lane = threadIdx.x & 31;
    if (wid >= (long)Nn * Hh) return;
    int n = (int)(wid / Hh), h = (int)(wid % Hh);
    const float* hp = H + (long)n * Hh * C + (long)h * C;
    const float* sp = a_s + h * C;
    const float* dp = a_d + h * C;
    float s = 0.f, d = 0.f;
    for (int c = lane; c < C; c += 32) { float v = hp[c]; s += v * sp[c]; d += v * dp[c]; }
    #pragma unroll
    for (int off = 16; off; off >>= 1) {
        s += __shfl_xor(s, off, 32);
        d += __shfl_xor(d, off, 32);
    }
    if (lane == 0) { als[wid] = s; ald[wid] = d; }
}

// ---------------- edge pass 1: e = leaky(als[src]+ald[dst], .2); segment max ----------------

__global__ void edge_max_kernel(const int* __restrict__ ei, int E, int Etot,
                                const float* __restrict__ als, const float* __restrict__ ald,
                                float* __restrict__ evals, unsigned* __restrict__ emax, int Hh) {
    int idx = blockIdx.x * blockDim.x + threadIdx.x;
    if (idx >= Etot * Hh) return;
    int e = idx / Hh, h = idx % Hh;
    int s, d;
    edge_nodes(ei, E, e, s, d);
    float v = lrelu(als[s * Hh + h] + ald[d * Hh + h], 0.2f);
    evals[idx] = v;
    atomicMax(&emax[d * Hh + h], fkey(v));
}

// ---------------- edge pass 2: den[dst,h] += exp(e - emax[dst,h]) ----------------

__global__ void edge_den_kernel(const int* __restrict__ ei, int E, int Etot,
                                const float* __restrict__ evals, const unsigned* __restrict__ emax,
                                float* __restrict__ den, int Hh) {
    int idx = blockIdx.x * blockDim.x + threadIdx.x;
    if (idx >= Etot * Hh) return;
    int e = idx / Hh, h = idx % Hh;
    int s, d;
    edge_nodes(ei, E, e, s, d);
    float m = fdec(emax[d * Hh + h]);
    atomicAdd(&den[d * Hh + h], expf(evals[idx] - m));
}

// ---------------- edge pass 3: Out[dst,h,:] += alpha * H[src,h,:], one wave per (edge,head) ---

__global__ __launch_bounds__(256)
void edge_msg_kernel(const int* __restrict__ ei, int E, int Etot,
                     const float* __restrict__ H, const float* __restrict__ evals,
                     const unsigned* __restrict__ emax, const float* __restrict__ den,
                     float* __restrict__ Out, int Hh, int C) {
    long wid = ((long)blockIdx.x * blockDim.x + threadIdx.x) >> 5;
    int lane = threadIdx.x & 31;
    if (wid >= (long)Etot * Hh) return;
    int e = (int)(wid / Hh), h = (int)(wid % Hh);
    int s, d;
    edge_nodes(ei, E, e, s, d);
    float m     = fdec(emax[d * Hh + h]);
    float ee    = expf(evals[wid] - m);
    float alpha = ee / (den[d * Hh + h] + 1e-16f);
    const float* hp = H + (long)s * Hh * C + (long)h * C;
    float*       op = Out + (long)d * Hh * C + (long)h * C;
    for (int c = lane; c < C; c += 32)
        atomicAdd(&op[c], hp[c] * alpha);
}

// ---------------- bias + leaky(0.01) ----------------

__global__ void bias_act_kernel(float* __restrict__ X, const float* __restrict__ b,
                                long total, int F) {
    long i = (long)blockIdx.x * blockDim.x + threadIdx.x;
    if (i >= total) return;
    X[i] = lrelu(X[i] + b[(int)(i % F)], 0.01f);
}

// ---------------- global mean pool + final GEMV ----------------

__global__ void pool_accum_kernel(const float* __restrict__ H, const int* __restrict__ batch,
                                  float* __restrict__ sums, float* __restrict__ cnt,
                                  int Nn, int F) {
    long idx = (long)blockIdx.x * blockDim.x + threadIdx.x;
    if (idx >= (long)Nn * F) return;
    int n = (int)(idx / F), c = (int)(idx % F);
    int g = batch[n];
    atomicAdd(&sums[(long)g * F + c], H[idx]);
    if (c == 0) atomicAdd(&cnt[g], 1.f);
}

__global__ void final_out_kernel(const float* __restrict__ sums, const float* __restrict__ cnt,
                                 const float* __restrict__ Wout, const float* __restrict__ bout,
                                 float* __restrict__ out, int F) {
    __shared__ float red[128];
    int g = blockIdx.x, c = threadIdx.x;
    float cn = cnt[g];
    cn = cn > 1.f ? cn : 1.f;
    red[c] = (sums[(long)g * F + c] / cn) * Wout[c];
    __syncthreads();
    for (int s = 64; s > 0; s >>= 1) {
        if (c < s) red[c] += red[c + s];
        __syncthreads();
    }
    if (c == 0) out[g] = red[0] + bout[0];
}

// ---------------- launcher ----------------

extern "C" void kernel_launch(void* const* d_in, const int* in_sizes, int n_in,
                              void* d_out, int out_size, void* d_ws, size_t ws_size,
                              hipStream_t stream) {
    const float* x     = (const float*)d_in[0];
    const int*   ei    = (const int*)d_in[1];
    const int*   batch = (const int*)d_in[2];
    const float* W1  = (const float*)d_in[3];
    const float* a1s = (const float*)d_in[4];
    const float* a1d = (const float*)d_in[5];
    const float* b1  = (const float*)d_in[6];
    const float* W2  = (const float*)d_in[7];
    const float* a2s = (const float*)d_in[8];
    const float* a2d = (const float*)d_in[9];
    const float* b2  = (const float*)d_in[10];
    const float* W3  = (const float*)d_in[11];
    const float* a3s = (const float*)d_in[12];
    const float* a3d = (const float*)d_in[13];
    const float* b3  = (const float*)d_in[14];
    const float* Wout = (const float*)d_in[15];
    const float* bout = (const float*)d_in[16];
    float* out = (float*)d_out;

    const int N    = in_sizes[0] / 11;
    const int E    = in_sizes[1] / 2;
    const int Etot = E + N;
    const int G = 64, FPOOL = 128;

    // workspace layout (f32 units)
    float* ws   = (float*)d_ws;
    size_t NA   = (size_t)N * 1024;
    float* bufA = ws;                 // GEMM output h (and L2 agg in its upper half)
    float* bufB = ws + NA;            // aggregated output / next-layer input
    float* als  = bufB + NA;          // [N, 8] max
    float* ald  = als + (size_t)N * 8;
    float* ev   = ald + (size_t)N * 8;            // [Etot, 8] max
    unsigned* emax = (unsigned*)(ev + (size_t)Etot * 8);  // [N, 8] keys
    float* den  = (float*)emax + (size_t)N * 8;
    float* sums = den + (size_t)N * 8;            // [64, 128]
    float* cnt  = sums + (size_t)G * FPOOL;       // [64]

    auto run_layer = [&](const float* Xin, const float* Wm, const float* a_s, const float* a_d,
                         const float* bias, float* Hbuf, float* Obuf, int Fin, int Hh, int C) {
        int Fout = Hh * C;   // always a multiple of 128
        dim3 gg(Fout / BN, (N + BM - 1) / BM);
        gemm_wmma_kernel<<<gg, 256, 0, stream>>>(Xin, Wm, Hbuf, N, Fin, Fout);

        long nh = (long)N * Hh;
        long at = nh * 32;  // one wave per (n,h)
        attn_logits_kernel<<<(unsigned)((at + 255) / 256), 256, 0, stream>>>(Hbuf, a_s, a_d, als, ald, N, Hh, C);

        hipMemsetAsync(emax, 0, (size_t)nh * sizeof(unsigned), stream);
        hipMemsetAsync(den, 0, (size_t)nh * sizeof(float), stream);

        long eh = (long)Etot * Hh;
        edge_max_kernel<<<(unsigned)((eh + 255) / 256), 256, 0, stream>>>(ei, E, Etot, als, ald, ev, emax, Hh);
        edge_den_kernel<<<(unsigned)((eh + 255) / 256), 256, 0, stream>>>(ei, E, Etot, ev, emax, den, Hh);

        hipMemsetAsync(Obuf, 0, (size_t)N * Fout * sizeof(float), stream);
        long threads = eh * 32;
        edge_msg_kernel<<<(unsigned)((threads + 255) / 256), 256, 0, stream>>>(
            ei, E, Etot, Hbuf, ev, emax, den, Obuf, Hh, C);

        long total = (long)N * Fout;
        bias_act_kernel<<<(unsigned)((total + 255) / 256), 256, 0, stream>>>(Obuf, bias, total, Fout);
    };

    // Layer 1: [N,11] -> h1 [N,1024] in bufA, agg -> bufB
    run_layer(x, W1, a1s, a1d, b1, bufA, bufB, 11, 8, 128);
    // Layer 2: [N,1024] -> h2 [N,512] in bufA[0..), agg -> bufA + N*512 (no overlap)
    run_layer(bufB, W2, a2s, a2d, b2, bufA, bufA + (size_t)N * 512, 1024, 8, 64);
    // Layer 3: [N,512] -> h3 [N,128] in bufB[0..), agg -> bufB + N*128
    run_layer(bufA + (size_t)N * 512, W3, a3s, a3d, b3, bufB, bufB + (size_t)N * 128, 512, 4, 32);

    // global mean pool + output projection
    const float* h3 = bufB + (size_t)N * 128;
    hipMemsetAsync(sums, 0, (size_t)G * FPOOL * sizeof(float), stream);
    hipMemsetAsync(cnt, 0, (size_t)G * sizeof(float), stream);
    long pthreads = (long)N * FPOOL;
    pool_accum_kernel<<<(unsigned)((pthreads + 255) / 256), 256, 0, stream>>>(h3, batch, sums, cnt, N, FPOOL);
    final_out_kernel<<<G, FPOOL, 0, stream>>>(sums, cnt, Wout, bout, out, FPOOL);
}
